// ComplexAttention_66099546685802
// MI455X (gfx1250) — compile-verified, matrix operand order checked by reading, not run on hardware
//
#include <hip/hip_runtime.h>
#include <hip/hip_bf16.h>

typedef __bf16 bf16_t;
typedef __attribute__((ext_vector_type(16))) __bf16 v16bf;
typedef __attribute__((ext_vector_type(8)))  __bf16 v8bf;
typedef __attribute__((ext_vector_type(8)))  float  v8f;
typedef __attribute__((ext_vector_type(4)))  int    v4i;

// ---------------------------------------------------------------------------
// CDNA5 async VMEM->LDS path (GLOBAL_LOAD_ASYNC_TO_LDS_B128, ASYNCcnt).
// Builtin signature (from toolchain diagnostic): (v4i addrspace(1)*,
// v4i addrspace(3)*, imm offset, imm cpol). Guarded by __has_builtin.
// ---------------------------------------------------------------------------
#if defined(__has_builtin)
#if __has_builtin(__builtin_amdgcn_global_load_async_to_lds_b128)
#define HAS_ASYNC_LDS 1
#endif
#endif

typedef __attribute__((address_space(1))) v4i gm_v4i_t;
typedef __attribute__((address_space(3))) v4i lds_v4i_t;

__device__ __forceinline__ void async_copy16(const void* g, void* l) {
#ifdef HAS_ASYNC_LDS
    // Flat LDS addr low 32 bits are the LDS byte offset (ISA 10.2 apertures).
    __builtin_amdgcn_global_load_async_to_lds_b128(
        (gm_v4i_t*)(uintptr_t)g, (lds_v4i_t*)(uint32_t)(uintptr_t)l, 0, 0);
#else
    *reinterpret_cast<v8bf*>(l) = *reinterpret_cast<const v8bf*>(g);
#endif
}

__device__ __forceinline__ void wait_async() {
#ifdef HAS_ASYNC_LDS
#if __has_builtin(__builtin_amdgcn_s_wait_asynccnt)
    __builtin_amdgcn_s_wait_asynccnt(0);
#else
    asm volatile("s_wait_asynccnt 0x0" ::: "memory");
#endif
#endif
}

__device__ __forceinline__ bf16_t us_to_bf(unsigned int u) {
    return __builtin_bit_cast(bf16_t, (unsigned short)u);
}

// ---------------------------------------------------------------------------
// WMMA helpers (CDNA5: v_wmma_f32_16x16x32_bf16, wave32)
// ---------------------------------------------------------------------------
__device__ __forceinline__ v8f wmma_bf16(v16bf a, v16bf b, v8f c) {
    return __builtin_amdgcn_wmma_f32_16x16x32_bf16(false, a, false, b,
                                                   (short)0, c, false, false);
}

// 16(row) x 32(k) bf16 operand tile in the A/B register layout:
// lane m (= lane&15) holds K = {0..7, 16..23}, lane m+16 holds K = {8..15, 24..31}.
__device__ __forceinline__ v16bf load_frag(const bf16_t* base, int strideHalf) {
    const int lane = threadIdx.x & 31;
    const int row  = lane & 15;
    const int ko   = (lane >> 4) << 3;          // 0 or 8
    const bf16_t* p = base + (size_t)row * strideHalf + ko;
    union { v16bf v; v8bf h[2]; } u;
    u.h[0] = *reinterpret_cast<const v8bf*>(p);
    u.h[1] = *reinterpret_cast<const v8bf*>(p + 16);
    return u.v;
}

// ---------------------------------------------------------------------------
// Prep kernels
// ---------------------------------------------------------------------------
__global__ void cvt_bf16_kernel(const float* __restrict__ src,
                                bf16_t* __restrict__ dst, size_t n) {
    size_t i = (size_t)blockIdx.x * blockDim.x + threadIdx.x;
    if (i < n) dst[i] = (bf16_t)src[i];
}

__global__ void pack_cplx_kernel(const float* __restrict__ wr,
                                 const float* __restrict__ wi,
                                 const float* __restrict__ br,
                                 const float* __restrict__ bi,
                                 bf16_t* __restrict__ Wp,
                                 float* __restrict__ Bp,
                                 int NO, int KI) {
    const int K2 = KI * 2;
    size_t total = (size_t)NO * 2 * K2;
    size_t idx = (size_t)blockIdx.x * blockDim.x + threadIdx.x;
    if (idx >= total) return;
    int r  = (int)(idx / K2);
    int kk = (int)(idx % K2);
    int n = r >> 1, c = r & 1, j = kk >> 1, p = kk & 1;
    float w;
    if (c == 0) w = (p == 0) ? wr[(size_t)n * KI + j] : -wi[(size_t)n * KI + j];
    else        w = (p == 0) ? wi[(size_t)n * KI + j] :  wr[(size_t)n * KI + j];
    Wp[idx] = (bf16_t)w;
    if (idx < (size_t)NO * 2) {
        int nn = (int)(idx >> 1);
        Bp[idx] = (idx & 1) ? (br[nn] + bi[nn]) : (br[nn] - bi[nn]);
    }
}

// ---------------------------------------------------------------------------
// Tiled bf16 WMMA GEMM:  C(MxN) = A(MxK) @ B(NxK)^T + bias[N]
// 128x128 block tile, K step 32, double-buffered async LDS staging.
// ---------------------------------------------------------------------------
#define GLDA 40   // padded LDS halfword stride (32 + 8) -> conflict-free b128 reads

template <bool OUT_BF16>
__global__ void __launch_bounds__(256)
gemm_bias_kernel(const bf16_t* __restrict__ A, const bf16_t* __restrict__ B,
                 const float* __restrict__ bias, void* __restrict__ Cout,
                 int M, int N, int K) {
    __shared__ __align__(16) bf16_t As[2][128 * GLDA];
    __shared__ __align__(16) bf16_t Bs[2][128 * GLDA];

    const int bm = blockIdx.y * 128;
    const int bn = blockIdx.x * 128;
    const int tid   = threadIdx.x;
    const int lane  = tid & 31;
    const int wave  = tid >> 5;
    const int waveM = wave & 3;
    const int waveN = wave >> 2;

    // Per-thread staging coords: 512 16B chunks per tile, 2 per thread.
    const int c0 = tid * 2,  r0 = c0 >> 2, p0 = (c0 & 3) << 3;
    const int c1 = c0 + 1,   r1 = c1 >> 2, p1 = (c1 & 3) << 3;

    auto stage = [&](int buf, int kb) {
        async_copy16(A + (size_t)(bm + r0) * K + kb + p0, &As[buf][r0 * GLDA + p0]);
        async_copy16(A + (size_t)(bm + r1) * K + kb + p1, &As[buf][r1 * GLDA + p1]);
        async_copy16(B + (size_t)(bn + r0) * K + kb + p0, &Bs[buf][r0 * GLDA + p0]);
        async_copy16(B + (size_t)(bn + r1) * K + kb + p1, &Bs[buf][r1 * GLDA + p1]);
    };

    v8f acc[2][4] = {};

    stage(0, 0);
    wait_async();
    __syncthreads();

    int buf = 0;
    for (int kb = 0; kb < K; kb += 32) {
        if (kb + 32 < K) stage(buf ^ 1, kb + 32);   // prefetch next block (async)

        v16bf af[2], bf[4];
#pragma unroll
        for (int mt = 0; mt < 2; ++mt)
            af[mt] = load_frag(&As[buf][(waveM * 32 + mt * 16) * GLDA], GLDA);
#pragma unroll
        for (int nt = 0; nt < 4; ++nt)
            bf[nt] = load_frag(&Bs[buf][(waveN * 64 + nt * 16) * GLDA], GLDA);
#pragma unroll
        for (int mt = 0; mt < 2; ++mt)
#pragma unroll
            for (int nt = 0; nt < 4; ++nt)
                acc[mt][nt] = wmma_bf16(af[mt], bf[nt], acc[mt][nt]);

        wait_async();
        __syncthreads();
        buf ^= 1;
    }

    const int hb = (lane >> 4) << 3;
#pragma unroll
    for (int mt = 0; mt < 2; ++mt) {
#pragma unroll
        for (int nt = 0; nt < 4; ++nt) {
            const int col = bn + waveN * 64 + nt * 16 + (lane & 15);
            const float bv = bias[col];
#pragma unroll
            for (int v = 0; v < 8; ++v) {
                const int row = bm + waveM * 32 + mt * 16 + v + hb;
                const float val = acc[mt][nt][v] + bv;
                if constexpr (OUT_BF16)
                    ((bf16_t*)Cout)[(size_t)row * N + col] = (bf16_t)val;
                else
                    ((float*)Cout)[(size_t)row * N + col] = val;
            }
        }
    }
}

// ---------------------------------------------------------------------------
// Flash attention per (b, h). qkv: 4096 x 6144 bf16 rows; head h -> 128 cols
// in each of the q/k/v thirds. 8 waves x 16 q rows per WG.
// S^T = K @ Q^T (q on lanes -> cheap row stats); P packs with zero cross-lane
// movement into the A layout for O += P @ V. K staged async, V staged
// transposed (global loads issued pre-compute, ds stores post-compute),
// double-buffered LDS.
// ---------------------------------------------------------------------------
#define ATT_N   2048
#define QKV_LD  6144
#define KLD     136   // Ks halfword stride (128 + 8 pad)
#define VLD     40    // Vs^T halfword stride (32 + 8 pad)

__global__ void __launch_bounds__(256)
attn_kernel(const bf16_t* __restrict__ qkv, bf16_t* __restrict__ out) {
    __shared__ __align__(16) bf16_t Ks[2][32 * KLD];   // 32 k-rows x 128
    __shared__ __align__(16) bf16_t Vs[2][128 * VLD];  // V^T: 128 d-rows x 32 k

    const int bh = blockIdx.x >> 4;
    const int qb = blockIdx.x & 15;
    const int b  = bh >> 4;                            // HEADS = 16
    const int h  = bh & 15;
    const int tid = threadIdx.x, wave = tid >> 5, lane = tid & 31;
    const int rowBase = b * ATT_N;
    const int qRow0   = qb * 128 + wave * 16;
    const float scale = 0.125f;                        // 1/sqrt(64)

    // K staging coords: 512 16B chunks, 2 per thread.
    const int kc0 = tid * 2, kr0 = kc0 >> 4, kp0 = (kc0 & 15) << 3;
    const int kc1 = kc0 + 1, kr1 = kc1 >> 4, kp1 = (kc1 & 15) << 3;
    // V staging coords: thread owns 8 contiguous dwords of one V row.
    const int vk  = (tid * 8) >> 6;                    // k row 0..31
    const int vj2 = ((tid * 8) & 63) << 1;             // d base (bf16 units)

    uint32_t vreg[8];

    auto k_stage = [&](int buf, int kb) {
        const bf16_t* kbase = qkv + 2048 + h * 128;
        async_copy16(kbase + (size_t)(rowBase + kb + kr0) * QKV_LD + kp0,
                     &Ks[buf][kr0 * KLD + kp0]);
        async_copy16(kbase + (size_t)(rowBase + kb + kr1) * QKV_LD + kp1,
                     &Ks[buf][kr1 * KLD + kp1]);
    };
    auto v_load = [&](int kb) {
        const uint32_t* src = reinterpret_cast<const uint32_t*>(
            qkv + (size_t)(rowBase + kb + vk) * QKV_LD + 4096 + h * 128 + vj2);
#pragma unroll
        for (int i = 0; i < 8; ++i) vreg[i] = src[i];
    };
    auto v_store = [&](int buf) {
#pragma unroll
        for (int i = 0; i < 8; ++i) {
            const int j2 = vj2 + 2 * i;
            Vs[buf][(j2 + 0) * VLD + vk] = us_to_bf(vreg[i] & 0xffffu);
            Vs[buf][(j2 + 1) * VLD + vk] = us_to_bf(vreg[i] >> 16);
        }
    };

    // Q operand (16 rows x 128 K) straight from global, register-resident.
    const bf16_t* qbase = qkv + (size_t)(rowBase + qRow0) * QKV_LD + h * 128;
    v16bf aq[4];
#pragma unroll
    for (int kc = 0; kc < 4; ++kc) aq[kc] = load_frag(qbase + kc * 32, QKV_LD);

    float m_run = -3.0e38f, l_run = 0.f;
    v8f Oc[8] = {};

    // Prologue stage of block 0.
    k_stage(0, 0);
    v_load(0);
    v_store(0);
    wait_async();
    __syncthreads();

    int buf = 0;
    for (int kb = 0; kb < ATT_N; kb += 32) {
        const bool more = (kb + 32) < ATT_N;
        if (more) {                       // prefetch next block before compute
            k_stage(buf ^ 1, kb + 32);    // async -> LDS
            v_load(kb + 32);              // global -> VGPRs (stored post-compute)
        }

        // ---- S^T tiles: two 16x16 C tiles (k on VGPRs, q on lanes) ----
        v8f st[2] = {};
#pragma unroll
        for (int t = 0; t < 2; ++t)
#pragma unroll
            for (int kc = 0; kc < 4; ++kc) {
                v16bf kf = load_frag(&Ks[buf][(t * 16) * KLD + kc * 32], KLD);
                st[t] = wmma_bf16(kf, aq[kc], st[t]);
            }

        // ---- online softmax (row = q = lane&15, dup across lane halves) ----
        float tmax = -3.0e38f;
#pragma unroll
        for (int t = 0; t < 2; ++t)
#pragma unroll
            for (int v = 0; v < 8; ++v) {
                float s = st[t][v] * scale;
                st[t][v] = s;
                tmax = fmaxf(tmax, s);
            }
        tmax = fmaxf(tmax, __shfl_xor(tmax, 16, 32));
        const float m_new = fmaxf(m_run, tmax);
        const float alpha = __expf(m_run - m_new);

        float psum = 0.f;
        union { v16bf v; v8bf h[2]; } ap;              // P in A-operand layout
#pragma unroll
        for (int t = 0; t < 2; ++t)
#pragma unroll
            for (int v = 0; v < 8; ++v) {
                float p = __expf(st[t][v] - m_new);
                psum += p;
                ap.v[t * 8 + v] = (bf16_t)p;
            }
        psum += __shfl_xor(psum, 16, 32);
        l_run = l_run * alpha + psum;
        m_run = m_new;

        // ---- rescale O (q across VGPRs) with per-row alpha via bpermute ----
        {
            const int hb = (lane >> 4) << 3;
#pragma unroll
            for (int v = 0; v < 8; ++v) {
                float a_v = __shfl(alpha, v + hb, 32);
#pragma unroll
                for (int td = 0; td < 8; ++td) Oc[td][v] *= a_v;
            }
        }
        // ---- O += P @ V over eight 16-wide d tiles ----
#pragma unroll
        for (int td = 0; td < 8; ++td) {
            v16bf vf = load_frag(&Vs[buf][(td * 16) * VLD], VLD);
            Oc[td] = wmma_bf16(ap.v, vf, Oc[td]);
        }

        if (more) v_store(buf ^ 1);       // transpose-store V after compute
        wait_async();
        __syncthreads();
        buf ^= 1;
    }

    // ---- epilogue: divide by l (per q-row), store bf16 for the out-proj GEMM
    const int hb = (lane >> 4) << 3;
    float linv[8];
#pragma unroll
    for (int v = 0; v < 8; ++v) linv[v] = 1.0f / __shfl(l_run, v + hb, 32);
#pragma unroll
    for (int td = 0; td < 8; ++td) {
        const int col = h * 128 + td * 16 + (lane & 15);
#pragma unroll
        for (int v = 0; v < 8; ++v) {
            const int row = rowBase + qRow0 + v + hb;
            out[(size_t)row * 2048 + col] = (bf16_t)(Oc[td][v] * linv[v]);
        }
    }
}

// ---------------------------------------------------------------------------
// Host launch
// ---------------------------------------------------------------------------
static inline int ceil_div(size_t a, int b) { return (int)((a + b - 1) / b); }

extern "C" void kernel_launch(void* const* d_in, const int* in_sizes, int n_in,
                              void* d_out, int out_size, void* d_ws, size_t ws_size,
                              hipStream_t stream) {
    (void)in_sizes; (void)n_in; (void)out_size; (void)ws_size;
    const float* x      = (const float*)d_in[0];
    const float* wqkv_r = (const float*)d_in[1];
    const float* wqkv_i = (const float*)d_in[2];
    const float* bqkv_r = (const float*)d_in[3];
    const float* bqkv_i = (const float*)d_in[4];
    const float* wo_r   = (const float*)d_in[5];
    const float* wo_i   = (const float*)d_in[6];
    const float* bo_r   = (const float*)d_in[7];
    const float* bo_i   = (const float*)d_in[8];
    float* out = (float*)d_out;

    // Workspace carve-up (bf16 halfwords), ~118 MB total.
    bf16_t* Xbf  = (bf16_t*)d_ws;                        // 4096 x 2048
    bf16_t* Wq   = Xbf  + (size_t)4096 * 2048;           // 6144 x 2048
    bf16_t* Wo   = Wq   + (size_t)6144 * 2048;           // 2048 x 2048
    bf16_t* QKV  = Wo   + (size_t)2048 * 2048;           // 4096 x 6144
    bf16_t* AOut = QKV  + (size_t)4096 * 6144;           // 4096 x 2048
    float*  Bq   = (float*)(AOut + (size_t)4096 * 2048); // 6144
    float*  Bo   = Bq + 6144;                            // 2048

    {
        size_t n = (size_t)4096 * 2048;
        cvt_bf16_kernel<<<ceil_div(n, 256), 256, 0, stream>>>(x, Xbf, n);
    }
    {
        size_t n = (size_t)6144 * 2048;
        pack_cplx_kernel<<<ceil_div(n, 256), 256, 0, stream>>>(
            wqkv_r, wqkv_i, bqkv_r, bqkv_i, Wq, Bq, 3072, 1024);
        size_t n2 = (size_t)2048 * 2048;
        pack_cplx_kernel<<<ceil_div(n2, 256), 256, 0, stream>>>(
            wo_r, wo_i, bo_r, bo_i, Wo, Bo, 1024, 1024);
    }
    // QKV projection: (4096x2048) @ (6144x2048)^T + bias -> bf16
    gemm_bias_kernel<true><<<dim3(6144 / 128, 4096 / 128), 256, 0, stream>>>(
        Xbf, Wq, Bq, QKV, 4096, 6144, 2048);
    // Flash attention per (b,h)
    attn_kernel<<<512, 256, 0, stream>>>(QKV, AOut);
    // Output projection: (4096x2048) @ (2048x2048)^T + bias -> fp32 d_out
    gemm_bias_kernel<false><<<dim3(2048 / 128, 4096 / 128), 256, 0, stream>>>(
        AOut, Wo, Bo, out, 4096, 2048, 2048);
}
